// VisualTokenSelection_79980880986198
// MI455X (gfx1250) — matrix operand order repaired and checked by reading, not compiled, not run on hardware
//
#include <hip/hip_runtime.h>
#include <hip/hip_bf16.h>

typedef __attribute__((ext_vector_type(16))) _Float16 v16h;
typedef __attribute__((ext_vector_type(8)))  float    v8f;

#define NFRAME   96      // B*MAX_FRAMES
#define NTOK     197
#define NROWS    (NFRAME*NTOK)   // 18912, divisible by 16
#define DIM      512
#define DH       256
#define NSPAT    196
#define TOPK_    49
#define NSAMP    256
#define SIGMA_   0.05f

union F16Frag { v16h v; uint4 q[2]; };

__device__ __forceinline__ float gelu_exact(float x) {
    return 0.5f * x * (1.0f + erff(x * 0.70710678118654752f));
}

// ---------------- Kernel 1: LayerNorm + f16 pack ----------------
__global__ void ln_kernel(const float* __restrict__ x,
                          const float* __restrict__ gamma,
                          const float* __restrict__ beta,
                          _Float16* __restrict__ out) {
    int row = blockIdx.x;          // 0..18911
    int t = threadIdx.x;           // 0..255
    const float* xr = x + (size_t)row * DIM;
    float a = xr[t], b = xr[t + 256];
    __shared__ float ss[256], sq[256];
    ss[t] = a + b;
    sq[t] = a * a + b * b;
    __syncthreads();
    for (int off = 128; off; off >>= 1) {
        if (t < off) { ss[t] += ss[t + off]; sq[t] += sq[t + off]; }
        __syncthreads();
    }
    float mu   = ss[0] * (1.0f / DIM);
    float var  = sq[0] * (1.0f / DIM) - mu * mu;
    float rstd = rsqrtf(var + 1e-5f);
    out[(size_t)row * DIM + t]       = (_Float16)((a - mu) * rstd * gamma[t]       + beta[t]);
    out[(size_t)row * DIM + t + 256] = (_Float16)((b - mu) * rstd * gamma[t + 256] + beta[t + 256]);
}

// ---------------- Kernel 2: transpose+convert weights to f16 [N][K] ----------------
__global__ void wcvt_kernel(const float* __restrict__ w_in,   // (512,256)
                            const float* __restrict__ w1,     // (512,256)
                            _Float16* __restrict__ wT,        // [256][512]
                            _Float16* __restrict__ w1T) {     // [256][256] (top half of w1)
    int idx = blockIdx.x * 256 + threadIdx.x;
    if (idx < DIM * DH) {
        int k = idx >> 8, n = idx & 255;                 // w_in[k][n]
        wT[(size_t)n * DIM + k] = (_Float16)w_in[idx];
    } else {
        int j = idx - DIM * DH;                          // 0..65535
        int k = j >> 8, n = j & 255;                     // w1[k][n], k<256
        w1T[(size_t)n * DH + k] = (_Float16)w1[j];
    }
}

// ---- WMMA fragment loaders (ISA 7.12.2 wave32 layouts) ----
__device__ __forceinline__ v16h load_afrag(const _Float16* __restrict__ arow,
                                           int kk, int sel) {
    // lanes 0-15: K{kk..kk+7, kk+16..kk+23}; lanes 16-31: K{kk+8..15, kk+24..31}
    F16Frag f;
    f.q[0] = *(const uint4*)(arow + kk + sel * 8);
    f.q[1] = *(const uint4*)(arow + kk + 16 + sel * 8);
    return f.v;
}
__device__ __forceinline__ v16h load_bfrag(const _Float16* __restrict__ brow,
                                           int kk, int sel) {
    // lanes 0-15: K{kk..kk+7, kk+16..kk+23}; lanes 16-31: K{kk+8..15, kk+24..31}
    F16Frag f;
    f.q[0] = *(const uint4*)(brow + kk + sel * 8);
    f.q[1] = *(const uint4*)(brow + kk + 16 + sel * 8);
    return f.v;
}

// 16x64-per-wave WMMA inner product over K (K = 512 or 256), all fragment
// loads hoisted into independent locals per iteration + next-A prefetch so
// the compiler can clause the loads and overlap their latency.
template <int K>
__device__ __forceinline__ void wmma_16x64(const _Float16* __restrict__ arow,
                                           const _Float16* __restrict__ b0,
                                           const _Float16* __restrict__ b1,
                                           const _Float16* __restrict__ b2,
                                           const _Float16* __restrict__ b3,
                                           int sel,
                                           v8f& acc0, v8f& acc1, v8f& acc2, v8f& acc3) {
    v16h av = load_afrag(arow, 0, sel);
#pragma unroll
    for (int kk = 0; kk < K; kk += 32) {
        v16h bv0 = load_bfrag(b0, kk, sel);
        v16h bv1 = load_bfrag(b1, kk, sel);
        v16h bv2 = load_bfrag(b2, kk, sel);
        v16h bv3 = load_bfrag(b3, kk, sel);
        v16h av_next = av;
        if (kk + 32 < K) av_next = load_afrag(arow, kk + 32, sel);  // prefetch streamed operand
        acc0 = __builtin_amdgcn_wmma_f32_16x16x32_f16(false, av, false, bv0, (short)0, acc0, false, false);
        acc1 = __builtin_amdgcn_wmma_f32_16x16x32_f16(false, av, false, bv1, (short)0, acc1, false, false);
        acc2 = __builtin_amdgcn_wmma_f32_16x16x32_f16(false, av, false, bv2, (short)0, acc2, false, false);
        acc3 = __builtin_amdgcn_wmma_f32_16x16x32_f16(false, av, false, bv3, (short)0, acc3, false, false);
        av = av_next;
    }
}

// ---------------- Kernel 3: h = gelu(hn @ w_in)  (f16 WMMA, 16x64 per wave) ----------------
__global__ void gemm1_kernel(const _Float16* __restrict__ A,   // [18912][512]
                             const _Float16* __restrict__ BT,  // [256][512]
                             _Float16* __restrict__ H) {       // [18912][256]
    int wid  = (blockIdx.x * blockDim.x + threadIdx.x) >> 5;   // global wave id
    int lane = threadIdx.x & 31;
    int nb     = wid & 3;            // 4 N-blocks of 64
    int tile_m = wid >> 2;           // 1182 M tiles
    int m0 = tile_m * 16, n0 = nb * 64;
    int laneH = lane & 15, sel = lane >> 4;
    const _Float16* arow = A  + (size_t)(m0 + laneH) * DIM;
    const _Float16* b0 = BT + (size_t)(n0      + laneH) * DIM;
    const _Float16* b1 = BT + (size_t)(n0 + 16 + laneH) * DIM;
    const _Float16* b2 = BT + (size_t)(n0 + 32 + laneH) * DIM;
    const _Float16* b3 = BT + (size_t)(n0 + 48 + laneH) * DIM;
    v8f acc0 = {}, acc1 = {}, acc2 = {}, acc3 = {};
    wmma_16x64<DIM>(arow, b0, b1, b2, b3, sel, acc0, acc1, acc2, acc3);
    for (int r = 0; r < 8; ++r) {
        size_t row = (size_t)(m0 + r + sel * 8) * DH;
        H[row + n0      + laneH] = (_Float16)gelu_exact(acc0[r]);
        H[row + n0 + 16 + laneH] = (_Float16)gelu_exact(acc1[r]);
        H[row + n0 + 32 + laneH] = (_Float16)gelu_exact(acc2[r]);
        H[row + n0 + 48 + laneH] = (_Float16)gelu_exact(acc3[r]);
    }
}

// ---------------- Kernel 4: c[f] = h_cls[f] @ w1_bot  (96x256) ----------------
__global__ void cvec_kernel(const _Float16* __restrict__ H,
                            const float* __restrict__ w1,
                            float* __restrict__ cvec) {
    int f = blockIdx.x, j = threadIdx.x;
    const _Float16* hc = H + (size_t)(f * NTOK) * DH;   // CLS row of frame f
    float acc = 0.f;
    for (int k = 0; k < DH; ++k)
        acc += (float)hc[k] * w1[(size_t)(DH + k) * DH + j];
    cvec[f * DH + j] = acc;
}

// ---------------- Kernel 5: s2 = gelu(h @ w1_top + c[frame])  (f16 WMMA, 16x64 per wave) ----------------
__global__ void gemm2_kernel(const _Float16* __restrict__ H,    // [18912][256]
                             const _Float16* __restrict__ W1T,  // [256][256]
                             const float* __restrict__ cvec,    // [96][256]
                             float* __restrict__ S2) {          // [18912][256]
    int wid  = (blockIdx.x * blockDim.x + threadIdx.x) >> 5;
    int lane = threadIdx.x & 31;
    int nb     = wid & 3;
    int tile_m = wid >> 2;
    int m0 = tile_m * 16, n0 = nb * 64;
    int laneH = lane & 15, sel = lane >> 4;
    const _Float16* arow = H   + (size_t)(m0 + laneH) * DH;
    const _Float16* b0 = W1T + (size_t)(n0      + laneH) * DH;
    const _Float16* b1 = W1T + (size_t)(n0 + 16 + laneH) * DH;
    const _Float16* b2 = W1T + (size_t)(n0 + 32 + laneH) * DH;
    const _Float16* b3 = W1T + (size_t)(n0 + 48 + laneH) * DH;
    v8f acc0 = {}, acc1 = {}, acc2 = {}, acc3 = {};
    wmma_16x64<DH>(arow, b0, b1, b2, b3, sel, acc0, acc1, acc2, acc3);
    for (int r = 0; r < 8; ++r) {
        int row = m0 + r + sel * 8;
        int f = row / NTOK;
        const float* cv = cvec + f * DH + n0 + laneH;
        float* so = S2 + (size_t)row * DH + n0 + laneH;
        so[0]  = gelu_exact(acc0[r] + cv[0]);
        so[16] = gelu_exact(acc1[r] + cv[16]);
        so[32] = gelu_exact(acc2[r] + cv[32]);
        so[48] = gelu_exact(acc3[r] + cv[48]);
    }
}

// ---------------- Kernel 6: score = tanh(s2 @ w2) ----------------
__global__ void score_kernel(const float* __restrict__ S2,
                             const float* __restrict__ w2,
                             float* __restrict__ score) {
    int row = blockIdx.x, t = threadIdx.x;
    __shared__ float sh[256];
    sh[t] = S2[(size_t)row * DH + t] * w2[t];
    __syncthreads();
    for (int off = 128; off; off >>= 1) {
        if (t < off) sh[t] += sh[t + off];
        __syncthreads();
    }
    if (t == 0) score[row] = tanhf(sh[0]);
}

// ---------------- Kernel 7: perturbed top-k selection, one wave32 per sample ----------------
__global__ void select_kernel(const float* __restrict__ score,
                              const float* __restrict__ noise,
                              float* __restrict__ cnt) {       // [96][49][196]
    int lane = threadIdx.x & 31;
    int wid  = threadIdx.x >> 5;
    int sg = blockIdx.x * 8 + wid;          // sample 0..24575
    int b  = sg >> 8;                        // frame 0..95
    const float* sp = score + b * NTOK + 1;  // 196 spatial scores
    const float* nz = noise + (size_t)sg * NSPAT;

    unsigned keys[7];
    for (int i = 0; i < 7; ++i) {
        int l = lane + i * 32;
        unsigned u = 0u;                     // invalid slots -> smallest key
        if (l < NSPAT) {
            float v = sp[l] + nz[l] * SIGMA_;
            unsigned bits = __float_as_uint(v);
            u = (bits & 0x80000000u) ? ~bits : (bits | 0x80000000u); // order-preserving
        }
        keys[i] = u;
    }

    // radix bit-descent: largest T with count(keys >= T) >= TOPK  ==  49th-largest key
    unsigned T = 0u;
    for (int bit = 31; bit >= 0; --bit) {
        unsigned cand = T | (1u << bit);
        int c = 0;
        for (int i = 0; i < 7; ++i) c += (keys[i] >= cand) ? 1 : 0;
        for (int off = 16; off; off >>= 1) c += __shfl_xor(c, off, 32);
        if (c >= TOPK_) T = cand;
    }

    int nGT = 0;
    for (int i = 0; i < 7; ++i) nGT += (keys[i] > T) ? 1 : 0;
    for (int off = 16; off; off >>= 1) nGT += __shfl_xor(nGT, off, 32);
    int needTies = TOPK_ - nGT;              // ties taken lowest-index-first (stable argsort)

    unsigned ltmask = (1u << lane) - 1u;
    int base = 0, eqbase = 0;
    for (int i = 0; i < 7; ++i) {
        bool eq = (keys[i] == T);
        unsigned em = (unsigned)__ballot(eq);
        int tierank = eqbase + __popc(em & ltmask);
        eqbase += __popc(em);
        bool selx = (keys[i] > T) || (eq && tierank < needTies);
        unsigned sm = (unsigned)__ballot(selx);
        int k = base + __popc(sm & ltmask);  // rank among selected, ascending index
        base += __popc(sm);
        int l = lane + i * 32;
        if (selx && l < NSPAT && k < TOPK_)
            atomicAdd(&cnt[((size_t)b * TOPK_ + k) * NSPAT + l], 1.0f);
    }
}

// ---------------- Kernel 8: out = [cls, (cnt/256) @ spat_feat] ----------------
__global__ void sel_kernel(const float* __restrict__ x,
                           const float* __restrict__ cnt,
                           float* __restrict__ out) {
    int bf = blockIdx.x / 50;
    int k  = blockIdx.x % 50;
    int t  = threadIdx.x;
    const float* xb = x + (size_t)bf * NTOK * DIM;
    float* o = out + ((size_t)bf * 50 + k) * DIM;
    if (k == 0) {                            // CLS passthrough
        o[t] = xb[t];
        o[t + 256] = xb[t + 256];
        return;
    }
    const float* cr = cnt + ((size_t)bf * TOPK_ + (k - 1)) * NSPAT;
    float a0 = 0.f, a1 = 0.f;
    for (int l = 0; l < NSPAT; ++l) {
        float c = cr[l];                     // uniform across block
        if (c != 0.0f) {
            const float* xr = xb + (size_t)(l + 1) * DIM;
            a0 += c * xr[t];
            a1 += c * xr[t + 256];
        }
    }
    o[t]       = a0 * (1.0f / NSAMP);
    o[t + 256] = a1 * (1.0f / NSAMP);
}

extern "C" void kernel_launch(void* const* d_in, const int* in_sizes, int n_in,
                              void* d_out, int out_size, void* d_ws, size_t ws_size,
                              hipStream_t stream) {
    const float* x       = (const float*)d_in[0];
    const float* noise   = (const float*)d_in[1];
    const float* g       = (const float*)d_in[2];
    const float* be      = (const float*)d_in[3];
    const float* w_in    = (const float*)d_in[4];
    const float* w1      = (const float*)d_in[5];
    const float* w2      = (const float*)d_in[6];
    float* out = (float*)d_out;

    char* ws = (char*)d_ws;
    size_t off = 0;
    _Float16* hn  = (_Float16*)(ws + off); off += (size_t)NROWS * DIM * 2;   // 19365888
    _Float16* wT  = (_Float16*)(ws + off); off += (size_t)DH * DIM * 2;      // 262144
    _Float16* w1T = (_Float16*)(ws + off); off += (size_t)DH * DH * 2;       // 131072
    _Float16* h   = (_Float16*)(ws + off); off += (size_t)NROWS * DH * 2;    // 9682944
    float* cvec   = (float*)(ws + off);    off += (size_t)NFRAME * DH * 4;   // 98304
    float* s2     = (float*)(ws + off);    off += (size_t)NROWS * DH * 4;    // 19365888
    float* score  = (float*)(ws + off);    off += ((size_t)NROWS * 4 + 255) & ~255ULL;
    float* cnt    = (float*)(ws + off);    off += (size_t)NFRAME * TOPK_ * NSPAT * 4;

    ln_kernel<<<NROWS, 256, 0, stream>>>(x, g, be, hn);
    wcvt_kernel<<<(DIM * DH + DH * DH) / 256, 256, 0, stream>>>(w_in, w1, wT, w1T);
    gemm1_kernel<<<(NROWS / 16) * 4 / 8, 256, 0, stream>>>(hn, wT, h);     // 591 blocks
    cvec_kernel<<<NFRAME, 256, 0, stream>>>(h, w1, cvec);
    gemm2_kernel<<<(NROWS / 16) * 4 / 8, 256, 0, stream>>>(h, w1T, cvec, s2);
    score_kernel<<<NROWS, 256, 0, stream>>>(s2, w2, score);
    hipMemsetAsync(cnt, 0, (size_t)NFRAME * TOPK_ * NSPAT * 4, stream);
    select_kernel<<<NFRAME * NSAMP / 8, 256, 0, stream>>>(score, noise, cnt);
    sel_kernel<<<NFRAME * 50, 256, 0, stream>>>(x, cnt, out);
}
